// RGCN_38104949850443
// MI455X (gfx1250) — matrix operand order repaired and checked by reading, not compiled, not run on hardware
//
#include <hip/hip_runtime.h>
#include <hip/hip_bf16.h>

typedef __attribute__((ext_vector_type(16))) __bf16 v16bf;
typedef __attribute__((ext_vector_type(8)))  float  v8f;

#define DIM 128   // D_IN == D_HID == D_OUT
#define WMAT_DWORD4 (DIM * DIM * 2 / 16)   // 2048 uint4 chunks per 128x128 bf16 matrix

// ---------------------------------------------------------------------------
// Load one 16x32 bf16 A-tile slice for this lane, per CDNA5 ISA layout:
//   lane L (L<16):  elems 0..7 -> K = 0..7,  elems 8..15 -> K = 16..23 (row L)
//   lane L (L>=16): elems 0..7 -> K = 8..15, elems 8..15 -> K = 24..31 (row L-16)
// 'p' must already point at x[row]*128 + kt*32 + kb  (kb = 0 or 8).
// ---------------------------------------------------------------------------
__device__ __forceinline__ v16bf load_a16x32(const float* __restrict__ p) {
    float4 f0 = *(const float4*)(p);
    float4 f1 = *(const float4*)(p + 4);
    float4 f2 = *(const float4*)(p + 16);
    float4 f3 = *(const float4*)(p + 20);
    v16bf a;
    a[0]=(__bf16)f0.x;  a[1]=(__bf16)f0.y;  a[2]=(__bf16)f0.z;  a[3]=(__bf16)f0.w;
    a[4]=(__bf16)f1.x;  a[5]=(__bf16)f1.y;  a[6]=(__bf16)f1.z;  a[7]=(__bf16)f1.w;
    a[8]=(__bf16)f2.x;  a[9]=(__bf16)f2.y;  a[10]=(__bf16)f2.z; a[11]=(__bf16)f2.w;
    a[12]=(__bf16)f3.x; a[13]=(__bf16)f3.y; a[14]=(__bf16)f3.z; a[15]=(__bf16)f3.w;
    return a;
}

// ---------------------------------------------------------------------------
// Pack f32 weight matrices [128x128] into bf16 in the exact per-lane WMMA
// B-matrix layout so each lane can load its 16 elements as 32 contiguous bytes:
//   packed[(((m*4 + kt)*8 + nt)*32 + lane)*16 + j] = W_m[kt*32 + kbase + (j&7) + (j>>3)*16][nt*16 + (lane&15)]
//   kbase = (lane<16) ? 0 : 8
// Matrices: m in [0,R) = W1_r ; m==R root1 ; m in [R+1,2R+1) = W2_r ; m==2R+1 root2
// ---------------------------------------------------------------------------
__global__ void pack_weights(const float* __restrict__ W1, const float* __restrict__ root1,
                             const float* __restrict__ W2, const float* __restrict__ root2,
                             __bf16* __restrict__ packed, int R) {
    int tid = blockIdx.x * blockDim.x + threadIdx.x;
    int total = (2 * R + 2) * DIM * DIM;
    if (tid >= total) return;
    int j    = tid & 15;
    int lane = (tid >> 4) & 31;
    int nt   = (tid >> 9) & 7;
    int kt   = (tid >> 12) & 3;
    int m    = tid >> 14;
    const float* src;
    if (m < R)              src = W1 + (size_t)m * DIM * DIM;
    else if (m == R)        src = root1;
    else if (m < 2 * R + 1) src = W2 + (size_t)(m - R - 1) * DIM * DIM;
    else                    src = root2;
    int col = nt * 16 + (lane & 15);
    int k   = kt * 32 + ((lane < 16) ? 0 : 8) + (j & 7) + ((j >> 3) << 4);
    packed[tid] = (__bf16)src[(size_t)k * DIM + col];
}

// ---------------------------------------------------------------------------
// Per-(dst, relation) in-degree + per-relation histogram.
// ---------------------------------------------------------------------------
__global__ void count_deg(const int* __restrict__ dstA, const int* __restrict__ et,
                          unsigned* __restrict__ cnt, unsigned* __restrict__ relHist,
                          int E, int R) {
    int e = blockIdx.x * blockDim.x + threadIdx.x;
    if (e >= E) return;
    int r = et[e];
    atomicAdd(&cnt[(size_t)dstA[e] * R + r], 1u);
    atomicAdd(&relHist[r], 1u);
}

// Tiny serial scan over R=8 bins: edge offsets + 16-edge tile offsets.
__global__ void scan_rel(const unsigned* __restrict__ relHist,
                         unsigned* __restrict__ binOff, unsigned* __restrict__ tileOff,
                         int R) {
    if (blockIdx.x == 0 && threadIdx.x == 0) {
        unsigned b = 0, t = 0;
        for (int r = 0; r < R; ++r) {
            binOff[r] = b; tileOff[r] = t;
            b += relHist[r];
            t += (relHist[r] + 15u) / 16u;
        }
        binOff[R] = b; tileOff[R] = t;
    }
}

// Counting-sort scatter: sorted[] holds edge ids grouped by relation.
__global__ void bin_edges(const int* __restrict__ et, const unsigned* __restrict__ binOff,
                          unsigned* __restrict__ cursor, unsigned* __restrict__ sorted,
                          int E) {
    int e = blockIdx.x * blockDim.x + threadIdx.x;
    if (e >= E) return;
    int r = et[e];
    unsigned p = atomicAdd(&cursor[r], 1u);
    sorted[binOff[r] + p] = e;
}

// ---------------------------------------------------------------------------
// Fused message kernel. Each block owns ONE relation (r = blockIdx.x % R) and
// stages that relation's packed 128x128 bf16 weight matrix in LDS (32 KB of
// the WGP's 320 KB). Each wave handles 16-edge tiles of that relation:
//   A (16x32 bf16)  = gathered x[src] rows, converted f32->bf16 (x is L2-hot)
//   B (32x16 bf16)  = ds_load_b128 x2 from LDS per (kt,nt) tile
//   D (16x128 f32)  = 8 accumulators, 4 k-steps * 8 n-tiles = 32 WMMAs/tile
// then rows are scaled by 1/max(deg(dst,r),1) and atomically added to agg.
// An opaque zero offset (inline asm) stops the compiler from promoting all
// 32 B-tiles to VGPRs (which caused scratch spills in the previous round).
// ---------------------------------------------------------------------------
__global__ void rgcn_edge_conv(const float* __restrict__ xin,
                               const __bf16* __restrict__ packedW,
                               const int* __restrict__ srcA,
                               const int* __restrict__ dstA,
                               const unsigned* __restrict__ sorted,
                               const unsigned* __restrict__ cnt,
                               const unsigned* __restrict__ binOff,
                               const unsigned* __restrict__ tileOff,
                               float* __restrict__ agg, int R) {
    __shared__ uint4 smemRaw[WMAT_DWORD4];
    const __bf16* smem = (const __bf16*)smemRaw;

    int r          = blockIdx.x % R;
    int blkInRel   = blockIdx.x / R;
    int blksPerRel = gridDim.x / R;

    // stage packed W_r into LDS
    {
        const uint4* gsrc = (const uint4*)(packedW + (size_t)r * DIM * DIM);
        for (int i = threadIdx.x; i < WMAT_DWORD4; i += blockDim.x)
            smemRaw[i] = gsrc[i];
    }
    __syncthreads();

    int lane        = threadIdx.x & 31;
    int wavesPerBlk = blockDim.x >> 5;
    int waveInBlk   = threadIdx.x >> 5;
    unsigned nTilesR = tileOff[r + 1] - tileOff[r];
    unsigned binBase = binOff[r];
    unsigned binEnd  = binOff[r + 1];
    unsigned stride  = (unsigned)(blksPerRel * wavesPerBlk);

    int rowA = lane & 15;
    int kb   = (lane < 16) ? 0 : 8;
    int half = (lane >> 4) << 3;
    int col0 = lane & 15;

    for (unsigned lt = (unsigned)(blkInRel * wavesPerBlk + waveInBlk); lt < nTilesR;
         lt += stride) {
        unsigned base = binBase + lt * 16u;

        // this lane's A-row source (rows beyond the bin load row 0; skipped on store)
        unsigned posA = base + (unsigned)rowA;
        int srcRow = 0;
        if (posA < binEnd) srcRow = srcA[sorted[posA]];
        const float* xp = xin + (size_t)srcRow * DIM + kb;

        // opaque zero: compiler cannot prove LDS reads loop-invariant -> keeps
        // ds_load_b128 per iteration instead of promoting 256 VGPRs + spilling
        unsigned sOff = 0;
        asm volatile("" : "+v"(sOff));
        const __bf16* sB = smem + sOff + (size_t)lane * 16;

        v8f acc[8] = {};
#pragma unroll
        for (int kt = 0; kt < 4; ++kt) {
            v16bf a = load_a16x32(xp + kt * 32);
#pragma unroll
            for (int nt = 0; nt < 8; ++nt) {
                v16bf b = *(const v16bf*)(sB + (size_t)(kt * 8 + nt) * 32 * 16);
                acc[nt] = __builtin_amdgcn_wmma_f32_16x16x32_bf16(
                    false, a, false, b, (short)0, acc[nt], false, false);
            }
        }

        // writeback: lane holds D rows M = half..half+7 at column col0 + 16*nt
        for (int v = 0; v < 8; ++v) {
            unsigned pos = base + (unsigned)(half + v);
            if (pos >= binEnd) continue;
            unsigned e = sorted[pos];
            int d = dstA[e];
            unsigned c = cnt[(size_t)d * R + r];
            float nrm = 1.0f / fmaxf((float)c, 1.0f);
            float* outp = agg + (size_t)d * DIM + col0;
#pragma unroll
            for (int nt = 0; nt < 8; ++nt)
                unsafeAtomicAdd(outp + nt * 16, acc[nt][v] * nrm);
        }
    }
}

// ---------------------------------------------------------------------------
// Root term: agg[node] += x[node] @ root + b  (self-loop "relation", norm=1).
// Root matrix staged in LDS; each node row owned by exactly one wave ->
// plain read-modify-write (edge kernel has already completed on the stream).
// ---------------------------------------------------------------------------
__global__ void rgcn_root(const float* __restrict__ xin,
                          const __bf16* __restrict__ packedRoot,
                          const float* __restrict__ bias,
                          float* __restrict__ agg, int nTiles, int N) {
    __shared__ uint4 smemRaw[WMAT_DWORD4];
    const __bf16* smem = (const __bf16*)smemRaw;
    {
        const uint4* gsrc = (const uint4*)packedRoot;
        for (int i = threadIdx.x; i < WMAT_DWORD4; i += blockDim.x)
            smemRaw[i] = gsrc[i];
    }
    __syncthreads();

    int lane        = threadIdx.x & 31;
    int wavesPerBlk = blockDim.x >> 5;
    int wave        = blockIdx.x * wavesPerBlk + (threadIdx.x >> 5);
    int nWaves      = gridDim.x * wavesPerBlk;

    int rowA = lane & 15;
    int kb   = (lane < 16) ? 0 : 8;
    int half = (lane >> 4) << 3;
    int col0 = lane & 15;

    for (int T = wave; T < nTiles; T += nWaves) {
        int node = T * 16 + rowA;
        int srcRow = (node < N) ? node : 0;
        const float* xp = xin + (size_t)srcRow * DIM + kb;

        unsigned sOff = 0;
        asm volatile("" : "+v"(sOff));
        const __bf16* sB = smem + sOff + (size_t)lane * 16;

        v8f acc[8] = {};
#pragma unroll
        for (int kt = 0; kt < 4; ++kt) {
            v16bf a = load_a16x32(xp + kt * 32);
#pragma unroll
            for (int nt = 0; nt < 8; ++nt) {
                v16bf b = *(const v16bf*)(sB + (size_t)(kt * 8 + nt) * 32 * 16);
                acc[nt] = __builtin_amdgcn_wmma_f32_16x16x32_bf16(
                    false, a, false, b, (short)0, acc[nt], false, false);
            }
        }

        for (int v = 0; v < 8; ++v) {
            int node2 = T * 16 + half + v;
            if (node2 >= N) continue;
            float* outp = agg + (size_t)node2 * DIM + col0;
#pragma unroll
            for (int nt = 0; nt < 8; ++nt)
                outp[nt * 16] = outp[nt * 16] + acc[nt][v] + bias[col0 + nt * 16];
        }
    }
}

__global__ void relu_inplace(float* __restrict__ h, size_t n) {
    size_t i = (size_t)blockIdx.x * blockDim.x + threadIdx.x;
    size_t stride = (size_t)gridDim.x * blockDim.x;
    for (; i < n; i += stride) h[i] = fmaxf(h[i], 0.0f);
}

// ---------------------------------------------------------------------------
// kernel_launch
// inputs: x, edge_index[2,E], edge_type[E], W1[R,128,128], root1, b1, W2, root2, b2
// ---------------------------------------------------------------------------
extern "C" void kernel_launch(void* const* d_in, const int* in_sizes, int n_in,
                              void* d_out, int out_size, void* d_ws, size_t ws_size,
                              hipStream_t stream) {
    const float* x     = (const float*)d_in[0];
    const int*   eidx  = (const int*)d_in[1];
    const int*   et    = (const int*)d_in[2];
    const float* W1    = (const float*)d_in[3];
    const float* root1 = (const float*)d_in[4];
    const float* b1    = (const float*)d_in[5];
    const float* W2    = (const float*)d_in[6];
    const float* root2 = (const float*)d_in[7];
    const float* b2    = (const float*)d_in[8];

    const int N = in_sizes[0] / DIM;
    const int E = in_sizes[2];
    const int R = in_sizes[3] / (DIM * DIM);
    const int* srcA = eidx;
    const int* dstA = eidx + E;

    // carve workspace (256B aligned regions)
    char* w = (char*)d_ws;
    auto carve = [&](size_t bytes) -> void* {
        void* p = (void*)w;
        w += (bytes + 255) & ~(size_t)255;
        return p;
    };
    unsigned* cnt     = (unsigned*)carve((size_t)N * R * sizeof(unsigned));
    unsigned* relHist = (unsigned*)carve((size_t)R * sizeof(unsigned));
    unsigned* binOff  = (unsigned*)carve((size_t)(R + 1) * sizeof(unsigned));
    unsigned* tileOff = (unsigned*)carve((size_t)(R + 1) * sizeof(unsigned));
    unsigned* cursor  = (unsigned*)carve((size_t)R * sizeof(unsigned));
    unsigned* sorted  = (unsigned*)carve((size_t)E * sizeof(unsigned));
    __bf16*   packed  = (__bf16*)carve((size_t)(2 * R + 2) * DIM * DIM * sizeof(__bf16));
    float*    h       = (float*)carve((size_t)N * DIM * sizeof(float));
    float*    out     = (float*)d_out;

    hipMemsetAsync(cnt, 0, (size_t)N * R * sizeof(unsigned), stream);
    hipMemsetAsync(relHist, 0, (size_t)R * sizeof(unsigned), stream);
    hipMemsetAsync(cursor, 0, (size_t)R * sizeof(unsigned), stream);
    hipMemsetAsync(h, 0, (size_t)N * DIM * sizeof(float), stream);
    hipMemsetAsync(out, 0, (size_t)out_size * sizeof(float), stream);

    // weight packing + graph preprocessing
    int packTotal = (2 * R + 2) * DIM * DIM;
    pack_weights<<<(packTotal + 255) / 256, 256, 0, stream>>>(W1, root1, W2, root2, packed, R);
    count_deg<<<(E + 255) / 256, 256, 0, stream>>>(dstA, et, cnt, relHist, E, R);
    scan_rel<<<1, 32, 0, stream>>>(relHist, binOff, tileOff, R);
    bin_edges<<<(E + 255) / 256, 256, 0, stream>>>(et, binOff, cursor, sorted, E);

    const int nTilesN    = (N + 15) / 16;
    const int convBlocks = 128 * R;   // blocksPerRel = 128, 4 waves each

    // layer 1: h = relu( sum_r mean-agg(x@W1_r) + x@root1 + b1 )
    rgcn_edge_conv<<<convBlocks, 128, 0, stream>>>(x, packed, srcA, dstA, sorted, cnt,
                                                   binOff, tileOff, h, R);
    rgcn_root<<<512, 128, 0, stream>>>(x, packed + (size_t)R * DIM * DIM, b1, h, nTilesN, N);
    relu_inplace<<<2048, 256, 0, stream>>>(h, (size_t)N * DIM);

    // layer 2: out = sum_r mean-agg(h@W2_r) + h@root2 + b2
    rgcn_edge_conv<<<convBlocks, 128, 0, stream>>>(h, packed + (size_t)(R + 1) * DIM * DIM,
                                                   srcA, dstA, sorted, cnt, binOff, tileOff,
                                                   out, R);
    rgcn_root<<<512, 128, 0, stream>>>(h, packed + (size_t)(2 * R + 1) * DIM * DIM, b2,
                                       out, nTilesN, N);
}
